// MarkovRegimeSwitchingLSTM_35416300322887
// MI455X (gfx1250) — compile-verified, matrix operand order checked by reading, not run on hardware
//
#include <hip/hip_runtime.h>
#include <hip/hip_bf16.h>

// ---------------------------------------------------------------------------
// Markov regime-switching LSTM for MI455X (gfx1250, wave32, WMMA).
//   B=64, S=256, IN=64, HS=512, K=4, OUT=32.
// Persistent kernel: 64 WGs (K=4 regimes x 16 hidden-slices of 32), weights
// live in LDS as pre-packed bf16 WMMA B-fragments, device-wide barrier per
// step, bf16 WMMA with fp32 accumulation. Inner loop is software-pipelined:
// next A fragment (global) prefetched across current 4 back-to-back WMMAs.
// ---------------------------------------------------------------------------

#define BATCH 64
#define SEQ   256
#define INP   64
#define HS    512
#define KREG  4
#define OUTD  32
#define GH    2048            // 4*HS
#define NWG   64              // persistent grid
#define TPB   256             // 8 waves
#define HSL   32              // hidden columns per WG
#define KT_V  16              // 512/32 k-tiles from V
#define KT_ALL 18             // + 2 k-tiles from U
#define NT    8               // 128/16 n-tiles
#define TILE_ELEMS 512        // 32 lanes * 16 bf16 per fragment tile
#define LDS_BYTES (KT_ALL * NT * TILE_ELEMS * 2)   // 147456 B = 144 KB

typedef __attribute__((ext_vector_type(16))) __bf16 v16bf;
typedef __attribute__((ext_vector_type(8)))  float  v8f;

#define WMMA_BF16(A, B, C) \
  __builtin_amdgcn_wmma_f32_16x16x32_bf16(false, (A), false, (B), (short)0, (C), false, false)

// ---- workspace layout (bytes) ----------------------------------------------
#define XFRAG_OFF 0u                      // S*8 tiles * 512 bf16 = 2 MB
#define HFRAG_OFF 2097152u                // 4*16 tiles * 512 bf16 = 64 KB
#define HCUR_OFF  2162688u                // 64*512 f32 = 128 KB
#define CCUR_OFF  2293760u                // 128 KB
#define HPART_OFF 2424832u                // K*64*512 f32 = 512 KB
#define CPART_OFF 2949120u                // 512 KB
#define BAR_OFF   3473408u                // 2 x u32

__device__ __forceinline__ unsigned short f2bf(float f) {
  unsigned u = __float_as_uint(f);
  unsigned r = u + 0x7FFFu + ((u >> 16) & 1u);   // round-to-nearest-even
  return (unsigned short)(r >> 16);
}

// ISA 16-bit A-matrix layout: lane group hi, element slot e -> K within 32
__device__ __forceinline__ int kprime(int hi, int e) {
  return (e & 7) | (hi << 3) | ((e & 8) << 1);
}

__device__ __forceinline__ float sigm(float x)  { return 1.0f / (1.0f + __expf(-x)); }
__device__ __forceinline__ float softplus5(float x) {
  float bx = 5.0f * x;
  return (bx > 5.0f) ? x : log1pf(__expf(fminf(bx, 5.0f))) * 0.2f;
}

__device__ __forceinline__ void grid_barrier(unsigned* cnt, unsigned* gen, unsigned nwg) {
  __threadfence();
  __syncthreads();
  if (threadIdx.x == 0) {
    unsigned g = __hip_atomic_load(gen, __ATOMIC_RELAXED, __HIP_MEMORY_SCOPE_AGENT);
    unsigned prev = __hip_atomic_fetch_add(cnt, 1u, __ATOMIC_ACQ_REL, __HIP_MEMORY_SCOPE_AGENT);
    if (prev == nwg - 1u) {
      __hip_atomic_store(cnt, 0u, __ATOMIC_RELAXED, __HIP_MEMORY_SCOPE_AGENT);
      __hip_atomic_fetch_add(gen, 1u, __ATOMIC_ACQ_REL, __HIP_MEMORY_SCOPE_AGENT);
    } else {
      while (__hip_atomic_load(gen, __ATOMIC_ACQUIRE, __HIP_MEMORY_SCOPE_AGENT) == g)
        __builtin_amdgcn_s_sleep(1);
    }
  }
  __syncthreads();
}

// ---------------------------------------------------------------------------
// Kernel 0: pack x into A-fragments for all timesteps; init h fragments,
// h/c state, and the barrier counters. Grid: SEQ+1 blocks x 256.
// ---------------------------------------------------------------------------
__global__ void __launch_bounds__(TPB) k_init(
    const float* __restrict__ x, const float* __restrict__ h0,
    const float* __restrict__ c0, unsigned short* __restrict__ xfrag,
    unsigned short* __restrict__ hfrag, float* __restrict__ hcur,
    float* __restrict__ ccur, unsigned* __restrict__ bar) {
  const int tid = threadIdx.x;
  if (blockIdx.x < SEQ) {
    const int t = blockIdx.x;
    // 8 tiles (4 m-blocks x 2 k-blocks) of 512 bf16 each
    for (int idx = tid; idx < 8 * TILE_ELEMS; idx += TPB) {
      int tile = idx >> 9, lane = (idx >> 4) & 31, e = idx & 15;
      int hi = lane >> 4;
      int kd = ((tile & 1) << 5) | kprime(hi, e);      // input feature 0..63
      int b  = ((tile >> 1) << 4) | (lane & 15);       // batch row
      xfrag[t * (8 * TILE_ELEMS) + idx] = f2bf(x[(b * SEQ + t) * INP + kd]);
    }
  } else {
    // h0 -> fragments (4 m-blocks x 16 k-blocks)
    for (int idx = tid; idx < 64 * TILE_ELEMS; idx += TPB) {
      int tile = idx >> 9, lane = (idx >> 4) & 31, e = idx & 15;
      int mi = tile >> 4, kt = tile & 15, hi = lane >> 4;
      int j = (kt << 5) | kprime(hi, e);
      int b = (mi << 4) | (lane & 15);
      hfrag[idx] = f2bf(h0[b * HS + j]);
    }
    for (int idx = tid; idx < BATCH * HS; idx += TPB) {
      hcur[idx] = h0[idx];
      ccur[idx] = c0[idx];
    }
    if (tid == 0) { bar[0] = 0u; bar[1] = 0u; }
  }
}

// ---------------------------------------------------------------------------
// Kernel 1: persistent LSTM. Grid: NWG x TPB, 144 KB dynamic LDS per WG.
// ---------------------------------------------------------------------------
__global__ void __launch_bounds__(TPB) k_lstm(
    const float* __restrict__ U, const float* __restrict__ V,
    const float* __restrict__ bvec, const float* __restrict__ alpha,
    const unsigned short* __restrict__ xfrag, unsigned short* __restrict__ hfrag,
    float* __restrict__ hcur, float* __restrict__ ccur,
    float* __restrict__ hpart, float* __restrict__ cpart,
    unsigned* __restrict__ bar) {
  extern __shared__ unsigned short ldsB[];   // [KT_ALL][NT][512] bf16 B-fragments

  const int tid  = threadIdx.x;
  const int lane = tid & 31;
  const int wave = tid >> 5;
  const int wg   = blockIdx.x;
  const int kr   = wg >> 4;                  // regime 0..3
  const int j0   = (wg & 15) * HSL;          // hidden slice base
  const int mi   = wave >> 1;                // batch block 0..3
  const int half = wave & 1;                 // hidden sub-half 0..1
  const int ncol = lane & 15;
  const int jj   = half * 16 + ncol;         // 0..31 within slice

  // ---- pack V slice into LDS B-fragments (kt 0..15) ----
  for (int idx = tid; idx < KT_V * NT * TILE_ELEMS; idx += TPB) {
    int kt = idx >> 12, rem = idx & 4095;
    int nt = rem >> 9, ln = (rem >> 4) & 31, e = idx & 15;
    int r  = (kt << 5) | kprime(ln >> 4, e);           // V row 0..511
    int g  = nt >> 1;
    int col = g * HS + j0 + ((nt & 1) << 4) + (ln & 15);
    ldsB[idx] = f2bf(V[((kr << 9) + r) * GH + col]);
  }
  // ---- pack U slice (kt 16..17) ----
  for (int idx = tid; idx < 2 * NT * TILE_ELEMS; idx += TPB) {
    int rem = idx & 4095;
    int nt = rem >> 9, ln = (rem >> 4) & 31, e = idx & 15;
    int r  = ((idx >> 12) << 5) | kprime(ln >> 4, e);  // U row 0..63
    int g  = nt >> 1;
    int col = g * HS + j0 + ((nt & 1) << 4) + (ln & 15);
    ldsB[KT_V * NT * TILE_ELEMS + idx] = f2bf(U[(kr * INP + r) * GH + col]);
  }
  // per-wave bias (one gate per accumulator; col depends on lane only)
  float bg0 = bvec[kr * GH + 0 * HS + j0 + jj];
  float bg1 = bvec[kr * GH + 1 * HS + j0 + jj];
  float bg2 = bvec[kr * GH + 2 * HS + j0 + jj];
  float bg3 = bvec[kr * GH + 3 * HS + j0 + jj];
  float a0r = alpha[0], a1r = alpha[1], a2r = alpha[2], a3r = alpha[3];

  const unsigned short* hbase = hfrag + ((mi * KT_V) * 32 + lane) * 16;
  const unsigned short* bb0   = ldsB + half * TILE_ELEMS + lane * 16;
  __syncthreads();

  #pragma unroll 1
  for (int t = 0; t < SEQ; ++t) {
    const unsigned short* xbase =
        xfrag + (size_t)t * (8 * TILE_ELEMS) + ((mi * 2) * 32 + lane) * 16;

    v8f acc0, acc1, acc2, acc3;
    #pragma unroll
    for (int v = 0; v < 8; ++v) { acc0[v] = bg0; acc1[v] = bg1; acc2[v] = bg2; acc3[v] = bg3; }

    // ---- gates = h @ V  (16 k-tiles, pipelined: prefetch next A fragment) --
    v16bf a = *(const v16bf*)hbase;
    #pragma unroll
    for (int kt = 0; kt < KT_V; ++kt) {
      const unsigned short* bb = bb0 + kt * (NT * TILE_ELEMS);
      v16bf b0 = *(const v16bf*)(bb);
      v16bf b1 = *(const v16bf*)(bb + 1024);
      v16bf b2 = *(const v16bf*)(bb + 2048);
      v16bf b3 = *(const v16bf*)(bb + 3072);
      v16bf an = (kt < KT_V - 1) ? *(const v16bf*)(hbase + (kt + 1) * TILE_ELEMS)
                                 : *(const v16bf*)(xbase);
      acc0 = WMMA_BF16(a, b0, acc0);   // i
      acc1 = WMMA_BF16(a, b1, acc1);   // f
      acc2 = WMMA_BF16(a, b2, acc2);   // g
      acc3 = WMMA_BF16(a, b3, acc3);   // o
      a = an;
    }
    // ---- gates += x_t @ U  (2 k-tiles) ------------------------------------
    #pragma unroll
    for (int kx = 0; kx < 2; ++kx) {
      const unsigned short* bb = bb0 + (KT_V + kx) * (NT * TILE_ELEMS);
      v16bf b0 = *(const v16bf*)(bb);
      v16bf b1 = *(const v16bf*)(bb + 1024);
      v16bf b2 = *(const v16bf*)(bb + 2048);
      v16bf b3 = *(const v16bf*)(bb + 3072);
      v16bf an = (kx == 0) ? *(const v16bf*)(xbase + TILE_ELEMS) : a;
      acc0 = WMMA_BF16(a, b0, acc0);
      acc1 = WMMA_BF16(a, b1, acc1);
      acc2 = WMMA_BF16(a, b2, acc2);
      acc3 = WMMA_BF16(a, b3, acc3);
      a = an;
    }

    // wave-local LSTM pointwise: this wave owns all 4 gates for its patch
    const int j = j0 + jj;
    #pragma unroll
    for (int v = 0; v < 8; ++v) {
      int m = (lane < 16) ? v : (v + 8);
      int b = mi * 16 + m;
      float it = sigm(acc0[v]);
      float ft = sigm(acc1[v]);
      float gt = tanhf(acc2[v]);
      float ot = sigm(acc3[v]);
      float ck = ft * ccur[b * HS + j] + it * gt;
      float hk = ot * tanhf(ck);
      cpart[(kr * BATCH + b) * HS + j] = ck;
      hpart[(kr * BATCH + b) * HS + j] = hk;
    }

    grid_barrier(bar, bar + 1, NWG);

    // regime mix: WGs of regime 0 reduce their hidden slice over K
    if (kr == 0) {
      for (int idx = tid; idx < BATCH * HSL; idx += TPB) {
        int b = idx >> 5, jl = idx & 31, jg = j0 + jl;
        float hn = a0r * hpart[(0 * BATCH + b) * HS + jg]
                 + a1r * hpart[(1 * BATCH + b) * HS + jg]
                 + a2r * hpart[(2 * BATCH + b) * HS + jg]
                 + a3r * hpart[(3 * BATCH + b) * HS + jg];
        float cn = a0r * cpart[(0 * BATCH + b) * HS + jg]
                 + a1r * cpart[(1 * BATCH + b) * HS + jg]
                 + a2r * cpart[(2 * BATCH + b) * HS + jg]
                 + a3r * cpart[(3 * BATCH + b) * HS + jg];
        hcur[b * HS + jg] = hn;
        ccur[b * HS + jg] = cn;
        // write h back in A-fragment order for next step's WMMA
        int kt = jg >> 5, kp = jg & 31;
        int hi = (kp >> 3) & 1;
        int e  = (kp & 7) | ((kp & 16) >> 1);
        int lb = (b & 15) | (hi << 4);
        hfrag[(((b >> 4) * KT_V + kt) * 32 + lb) * 16 + e] = f2bf(hn);
      }
    }

    grid_barrier(bar, bar + 1, NWG);
  }
  // after loop: hcur/ccur hold h_T/c_T; hpart holds raw per-regime h_k at t=S-1
}

// ---------------------------------------------------------------------------
// Kernel 2: output head. out = softplus5(h_T @ W_out); y_tk from hpart;
// plus state copies. Grid: 64 x 256.
// ---------------------------------------------------------------------------
__global__ void __launch_bounds__(TPB) k_final(
    const float* __restrict__ hcur, const float* __restrict__ ccur,
    const float* __restrict__ hpart, const float* __restrict__ Wout,
    const float* __restrict__ alpha, float* __restrict__ out) {
  const int gid = blockIdx.x * TPB + threadIdx.x;
  const int nthreads = 64 * TPB;

  if (gid < (BATCH + KREG * BATCH) * OUTD) {        // 320 rows x 32 cols
    int row = gid >> 5, o = gid & 31;
    const float* src = (row < BATCH) ? (hcur + row * HS)
                                     : (hpart + (size_t)(row - BATCH) * HS);
    float s = 0.0f;
    for (int jv = 0; jv < HS; ++jv) s += src[jv] * Wout[jv * OUTD + o];
    float r = softplus5(s);
    if (row < BATCH) out[row * OUTD + o] = r;                       // out
    else out[2048 + 2 * BATCH * HS + (row - BATCH) * OUTD + o] = r; // y_tk
  }
  for (int idx = gid; idx < 2 * BATCH * HS; idx += nthreads) {      // h_t, c_t
    out[2048 + idx] = (idx < BATCH * HS) ? hcur[idx] : ccur[idx - BATCH * HS];
  }
  if (gid < KREG) out[2048 + 2 * BATCH * HS + KREG * BATCH * OUTD + gid] = alpha[gid];
}

// ---------------------------------------------------------------------------
extern "C" void kernel_launch(void* const* d_in, const int* in_sizes, int n_in,
                              void* d_out, int out_size, void* d_ws, size_t ws_size,
                              hipStream_t stream) {
  (void)in_sizes; (void)n_in; (void)out_size; (void)ws_size;
  const float* x     = (const float*)d_in[0];
  const float* h0    = (const float*)d_in[1];
  const float* c0    = (const float*)d_in[2];
  const float* alpha = (const float*)d_in[3];
  const float* U     = (const float*)d_in[4];
  const float* V     = (const float*)d_in[5];
  const float* bvec  = (const float*)d_in[6];
  const float* Wout  = (const float*)d_in[7];
  float* out = (float*)d_out;

  char* ws = (char*)d_ws;
  unsigned short* xfrag = (unsigned short*)(ws + XFRAG_OFF);
  unsigned short* hfrag = (unsigned short*)(ws + HFRAG_OFF);
  float* hcur  = (float*)(ws + HCUR_OFF);
  float* ccur  = (float*)(ws + CCUR_OFF);
  float* hpart = (float*)(ws + HPART_OFF);
  float* cpart = (float*)(ws + CPART_OFF);
  unsigned* bar = (unsigned*)(ws + BAR_OFF);

  // allow 144 KB dynamic LDS (gfx1250 WGP has 320 KB); host-side, capture-safe
  hipFuncSetAttribute((const void*)k_lstm,
                      hipFuncAttributeMaxDynamicSharedMemorySize, LDS_BYTES);

  k_init<<<SEQ + 1, TPB, 0, stream>>>(x, h0, c0, xfrag, hfrag, hcur, ccur, bar);
  k_lstm<<<NWG, TPB, LDS_BYTES, stream>>>(U, V, bvec, alpha, xfrag, hfrag,
                                          hcur, ccur, hpart, cpart, bar);
  k_final<<<64, TPB, 0, stream>>>(hcur, ccur, hpart, Wout, alpha, out);
}
